// DissipativeThetaRINN_20933670600777
// MI455X (gfx1250) — matrix operand order repaired and checked by reading, not compiled, not run on hardware
//
#include <hip/hip_runtime.h>

// ---------------------------------------------------------------------------
// Types
// ---------------------------------------------------------------------------
typedef __attribute__((ext_vector_type(16))) __bf16 bf16x16;
typedef __attribute__((ext_vector_type(2)))  __bf16 bf16x2;
typedef __attribute__((ext_vector_type(8)))  float  f32x8;

union FragB { unsigned int u[8]; bf16x16 v; };

static __device__ __forceinline__ f32x8 zero8() {
  f32x8 z;
#pragma unroll
  for (int i = 0; i < 8; ++i) z[i] = 0.0f;
  return z;
}

static __device__ __forceinline__ f32x8 wmma_bf16(bf16x16 a, bf16x16 b, f32x8 c) {
  // D = A(16x32 bf16) * B(32x16 bf16) + C(16x16 f32)
  return __builtin_amdgcn_wmma_f32_16x16x32_bf16(false, a, false, b, (short)0, c,
                                                 false, false);
}

// ---- tanh: prefer native builtin (compiler handles trans hazards) ---------
static __device__ __forceinline__ float fast_tanh(float x) {
#if __has_builtin(__builtin_amdgcn_tanhf)
  return __builtin_amdgcn_tanhf(x);
#else
  float r;
  asm("v_tanh_f32 %0, %1\n\tv_nop" : "=v"(r) : "v"(x));
  return r;
#endif
}

// ---- f32 -> bf16: prefer v_cvt_pk_bf16_f32 --------------------------------
#if __has_builtin(__builtin_amdgcn_cvt_pk_bf16_f32)
static __device__ __forceinline__ unsigned int pack_bf16(float a, float b) {
  union { bf16x2 v; unsigned int u; } cv;
  cv.v = __builtin_amdgcn_cvt_pk_bf16_f32(a, b);  // D[15:0]=a, D[31:16]=b
  return cv.u;
}
static __device__ __forceinline__ unsigned short f32_to_bf16(float f) {
  return (unsigned short)(pack_bf16(f, 0.0f) & 0xFFFFu);
}
#else
// round-half-up (cheap fallback, 2 VALU ops)
static __device__ __forceinline__ unsigned short f32_to_bf16(float f) {
  unsigned int u = __float_as_uint(f) + 0x8000u;
  return (unsigned short)(u >> 16);
}
static __device__ __forceinline__ unsigned int pack_bf16(float a, float b) {
  return (unsigned int)f32_to_bf16(a) | ((unsigned int)f32_to_bf16(b) << 16);
}
#endif

// Build a B-matrix fragment (32x16 bf16) from an arbitrary getter g(k, n).
// Layout: lanes 0-15 hold K=0..7 (V0-3) and K=16..23 (V4-7), lanes 16-31 hold +8.
template <typename F>
static __device__ __forceinline__ bf16x16 build_bfrag(int lane, int kchunk,
                                                      int ntile, F g) {
  FragB f;
  const int n  = ntile * 16 + (lane & 15);
  const int kb = kchunk * 32 + ((lane & 16) ? 8 : 0);
#pragma unroll
  for (int v = 0; v < 8; ++v) {
    const int k0 = kb + 2 * v + (v >= 4 ? 8 : 0);
    f.u[v] = pack_bf16(g(k0, n), g(k0 + 1, n));
  }
  return f.v;
}

// Load an A-matrix fragment (16x32 bf16) from row-major bf16 LDS tile.
// Lanes 0-15: row m=lane, halves K=kofs+{0..7} (V0-3) and K=kofs+{16..23} (V4-7).
// Lanes 16-31: row m=lane-16, halves +8.
static __device__ __forceinline__ bf16x16 lds_afrag(const unsigned short* base,
                                                    int strideH, int kofs,
                                                    int lane) {
  const int m   = lane & 15;
  const int kb8 = (lane & 16) ? 8 : 0;
  const uint4* q =
      reinterpret_cast<const uint4*>(base + m * strideH + kofs + kb8);
  FragB f;
  uint4 lo = q[0];   // halves kofs+kb8 .. +7
  uint4 hi = q[2];   // halves kofs+kb8+16 .. +23
  f.u[0] = lo.x; f.u[1] = lo.y; f.u[2] = lo.z; f.u[3] = lo.w;
  f.u[4] = hi.x; f.u[5] = hi.y; f.u[6] = hi.z; f.u[7] = hi.w;
  return f.v;
}

// Store a 16x16 f32 D tile (C/D layout) as bf16 into row-major LDS.
// lanes 0-15: N=lane, M=v ; lanes 16-31: N=lane-16, M=v+8.
static __device__ __forceinline__ void lds_store_tile(unsigned short* base,
                                                      int strideH, int colbase,
                                                      f32x8 d, int lane) {
  const int mb = (lane & 16) ? 8 : 0;
  const int c  = colbase + (lane & 15);
#pragma unroll
  for (int v = 0; v < 8; ++v) base[(v + mb) * strideH + c] = f32_to_bf16(d[v]);
}

// ---------------------------------------------------------------------------
// Policy recurrence kernel
//   1 workgroup (8 waves, 256 thr) per 16 batch rows; wave i owns w N-tile i.
// ---------------------------------------------------------------------------
#define S_   16
#define NL_  128
#define IN_  32
#define OUT_ 8
#define T_   128
#define OC_  17
#define DT_  0.01f
#define FPI_ 30

__global__ __launch_bounds__(256) void rinn_policy_kernel(
    const float* __restrict__ obs, const float* __restrict__ x0,
    const float* __restrict__ A_T, const float* __restrict__ Bw_T,
    const float* __restrict__ By_T, const float* __restrict__ Cv_T,
    const float* __restrict__ Dvw_T, const float* __restrict__ Dvy_T,
    const float* __restrict__ Cu_T, const float* __restrict__ Duw_T,
    const float* __restrict__ Duy_T, float* __restrict__ out) {
  __shared__ __align__(16) unsigned short sXY[16 * 64];        // z = [x|y|pad]
  __shared__ __align__(16) unsigned short sW[2][16 * NL_];     // double-buffered w

  const int tid  = threadIdx.x;
  const int lane = tid & 31;
  const int wave = tid >> 5;
  const int r0   = blockIdx.x * 16;  // batch row base

  // ---- loop-invariant weight fragments in registers -----------------------
  bf16x16 dvwf[4];  // Dvw B-frags for this wave's N-tile
#pragma unroll
  for (int c = 0; c < 4; ++c)
    dvwf[c] = build_bfrag(lane, c, wave,
                          [&](int k, int n) { return Dvw_T[k * NL_ + n]; });

  // E[64][128]: rows 0-15 Cv_T, rows 16-47 Dvy_T, rows 48-63 zero
  bf16x16 ef[2];
#pragma unroll
  for (int c = 0; c < 2; ++c)
    ef[c] = build_bfrag(lane, c, wave, [&](int k, int n) {
      if (k < 16) return Cv_T[k * NL_ + n];
      if (k < 48) return Dvy_T[(k - 16) * NL_ + n];
      return 0.0f;
    });

  // H[192][32]: cols 0-7 -> u weights, cols 16-31 -> xdelta weights
  bf16x16 hf[6];
  if (wave < 2) {
#pragma unroll
    for (int c = 0; c < 6; ++c)
      hf[c] = build_bfrag(lane, c, wave, [&](int k, int n) {
        float g = 0.0f;
        if (n < 8) {
          if (k < 16)       g = Cu_T[k * OUT_ + n];
          else if (k < 48)  g = Duy_T[(k - 16) * OUT_ + n];
          else if (k >= 64) g = Duw_T[(k - 64) * OUT_ + n];
        } else if (n >= 16) {
          const int s = n - 16;
          if (k < 16)       g = A_T[k * S_ + s];
          else if (k < 48)  g = By_T[(k - 16) * S_ + s];
          else if (k >= 64) g = Bw_T[(k - 64) * S_ + s];
        }
        return g;
      });
  }

  // ---- state x in f32 D-layout registers (wave 1) -------------------------
  f32x8 xD = zero8();
  const int mb = (lane & 16) ? 8 : 0;
  if (wave == 1) {
#pragma unroll
    for (int v = 0; v < 8; ++v)
      xD[v] = x0[(r0 + v + mb) * S_ + (lane & 15)];
  }

  // zero sXY (incl. pad cols 48-63)
#pragma unroll
  for (int i = tid; i < 16 * 64; i += 256) sXY[i] = 0;
  __syncthreads();

  for (int t = 0; t < T_; ++t) {
    // stage y = obs[:, t, :] as bf16 into cols 16..47
#pragma unroll
    for (int e = tid; e < 16 * IN_; e += 256) {
      const int m = e >> 5, i = e & 31;
      sXY[m * 64 + 16 + i] =
          f32_to_bf16(obs[((size_t)(r0 + m) * T_ + t) * IN_ + i]);
    }
    // stage x as bf16 into cols 0..15
    if (wave == 1) {
#pragma unroll
      for (int v = 0; v < 8; ++v)
        sXY[(v + mb) * 64 + (lane & 15)] = f32_to_bf16(xD[v]);
    }
    __syncthreads();

    // const = [x|y] @ E   (this wave's N-tile, f32 regs)
    bf16x16 axy0 = lds_afrag(sXY, 64, 0, lane);
    bf16x16 axy1 = lds_afrag(sXY, 64, 32, lane);
    f32x8 cst = wmma_bf16(axy0, ef[0], zero8());
    cst       = wmma_bf16(axy1, ef[1], cst);

    // w1 = tanh(const)
    f32x8 wv;
#pragma unroll
    for (int j = 0; j < 8; ++j) wv[j] = fast_tanh(cst[j]);
    lds_store_tile(sW[0], NL_, wave * 16, wv, lane);
    __syncthreads();

    // iterations 2..30:  w = tanh(const + w @ Dvw)
    for (int it = 1; it < FPI_; ++it) {
      const unsigned short* rd = sW[(it - 1) & 1];
      bf16x16 wa[4];
#pragma unroll
      for (int c = 0; c < 4; ++c) wa[c] = lds_afrag(rd, NL_, c * 32, lane);
      f32x8 acc = cst;
#pragma unroll
      for (int c = 0; c < 4; ++c) acc = wmma_bf16(wa[c], dvwf[c], acc);
#pragma unroll
      for (int j = 0; j < 8; ++j) wv[j] = fast_tanh(acc[j]);
      lds_store_tile(sW[it & 1], NL_, wave * 16, wv, lane);
      __syncthreads();
    }

    // outputs: wave0 -> u, wave1 -> x update   (final w is in sW[1])
    if (wave < 2) {
      bf16x16 wa[4];
#pragma unroll
      for (int c = 0; c < 4; ++c) wa[c] = lds_afrag(sW[1], NL_, c * 32, lane);
      f32x8 d = wmma_bf16(axy0, hf[0], zero8());
      d       = wmma_bf16(axy1, hf[1], d);
#pragma unroll
      for (int c = 0; c < 4; ++c) d = wmma_bf16(wa[c], hf[2 + c], d);
      if (wave == 0) {
        const int n = lane & 15;
        if (n < 8) {
#pragma unroll
          for (int v = 0; v < 8; ++v)
            out[((size_t)(r0 + v + mb) * T_ + t) * OC_ + n] = d[v];
        }
      } else {
#pragma unroll
        for (int v = 0; v < 8; ++v) xD[v] += DT_ * d[v];
      }
    }
    __syncthreads();
  }
}

// ---------------------------------------------------------------------------
// Value branch + log_stds broadcast:  per wave, 16 (b,t) rows.
// ---------------------------------------------------------------------------
__global__ __launch_bounds__(256) void rinn_value_kernel(
    const float* __restrict__ obs, const float* __restrict__ log_stds,
    const float* __restrict__ W0, const float* __restrict__ b0,
    const float* __restrict__ W1, const float* __restrict__ b1,
    const float* __restrict__ W2, const float* __restrict__ b2,
    float* __restrict__ out) {
  __shared__ __align__(16) unsigned short sH[8][16 * 64];  // bf16 h round-trip
  __shared__ float sV[8][16 * 68];                         // padded f32 h2

  const int tid  = threadIdx.x;
  const int lane = tid & 31;
  const int wave = tid >> 5;
  const long r0  = ((long)blockIdx.x * 8 + wave) * 16;  // flat (b*T+t) row base

  // weight fragments in registers
  bf16x16 w0f[4];
#pragma unroll
  for (int nt = 0; nt < 4; ++nt)
    w0f[nt] = build_bfrag(lane, 0, nt,
                          [&](int k, int n) { return W0[k * 64 + n]; });
  bf16x16 w1f[2][4];
#pragma unroll
  for (int c = 0; c < 2; ++c)
#pragma unroll
    for (int nt = 0; nt < 4; ++nt)
      w1f[c][nt] = build_bfrag(lane, c, nt,
                               [&](int k, int n) { return W1[k * 64 + n]; });
  float b0v[4], b1v[4];
#pragma unroll
  for (int nt = 0; nt < 4; ++nt) {
    b0v[nt] = b0[nt * 16 + (lane & 15)];
    b1v[nt] = b1[nt * 16 + (lane & 15)];
  }

  // obs A-fragment (16 rows x 32 bf16) gathered from global f32
  const int m   = lane & 15;
  const int kb8 = (lane & 16) ? 8 : 0;
  FragB fa;
#pragma unroll
  for (int j = 0; j < 4; ++j) {
    const float* p = obs + (r0 + m) * IN_;
    fa.u[j]     = pack_bf16(p[kb8 + 2 * j], p[kb8 + 2 * j + 1]);
    fa.u[4 + j] = pack_bf16(p[kb8 + 16 + 2 * j], p[kb8 + 16 + 2 * j + 1]);
  }
  bf16x16 aob = fa.v;

  const int mb = (lane & 16) ? 8 : 0;
  // layer 1: h = tanh(obs @ W0 + b0)
#pragma unroll
  for (int nt = 0; nt < 4; ++nt) {
    f32x8 d = wmma_bf16(aob, w0f[nt], zero8());
#pragma unroll
    for (int v = 0; v < 8; ++v)
      sH[wave][(v + mb) * 64 + nt * 16 + (lane & 15)] =
          f32_to_bf16(fast_tanh(d[v] + b0v[nt]));
  }

  bf16x16 ha0 = lds_afrag(sH[wave], 64, 0, lane);
  bf16x16 ha1 = lds_afrag(sH[wave], 64, 32, lane);

  // layer 2: h2 = tanh(h @ W1 + b1)
#pragma unroll
  for (int nt = 0; nt < 4; ++nt) {
    f32x8 d = wmma_bf16(ha0, w1f[0][nt], zero8());
    d       = wmma_bf16(ha1, w1f[1][nt], d);
#pragma unroll
    for (int v = 0; v < 8; ++v)
      sV[wave][(v + mb) * 68 + nt * 16 + (lane & 15)] =
          fast_tanh(d[v] + b1v[nt]);
  }
  // wave-local LDS RAW fence (stores above, loads below, same wave)
  asm volatile("s_wait_dscnt 0x0" ::: "memory");

  // value = h2 @ W2 + b2 ; also write log_stds broadcast (cols 8..16)
  if (lane < 16) {
    float acc = b2[0];
#pragma unroll
    for (int k = 0; k < 64; ++k) acc += sV[wave][lane * 68 + k] * W2[k];
    float* o = out + (r0 + lane) * OC_;
#pragma unroll
    for (int j = 0; j < OUT_; ++j) o[OUT_ + j] = log_stds[j];
    o[2 * OUT_] = acc;
  }
}

// ---------------------------------------------------------------------------
// Launch
// ---------------------------------------------------------------------------
extern "C" void kernel_launch(void* const* d_in, const int* in_sizes, int n_in,
                              void* d_out, int out_size, void* d_ws,
                              size_t ws_size, hipStream_t stream) {
  const float* obs      = (const float*)d_in[0];
  const float* x0       = (const float*)d_in[1];
  const float* A_T      = (const float*)d_in[2];
  const float* Bw_T     = (const float*)d_in[3];
  const float* By_T     = (const float*)d_in[4];
  const float* Cv_T     = (const float*)d_in[5];
  const float* Dvw_T    = (const float*)d_in[6];
  const float* Dvy_T    = (const float*)d_in[7];
  const float* Cu_T     = (const float*)d_in[8];
  const float* Duw_T    = (const float*)d_in[9];
  const float* Duy_T    = (const float*)d_in[10];
  const float* log_stds = (const float*)d_in[11];
  const float* W0       = (const float*)d_in[12];
  const float* b0       = (const float*)d_in[13];
  const float* W1       = (const float*)d_in[14];
  const float* b1       = (const float*)d_in[15];
  const float* W2       = (const float*)d_in[16];
  const float* b2       = (const float*)d_in[17];
  float* out            = (float*)d_out;

  const int B  = in_sizes[1] / S_;          // x0 is [B, S]
  const int BT = in_sizes[0] / IN_;         // obs is [B*T, IN]

  rinn_policy_kernel<<<B / 16, 256, 0, stream>>>(
      obs, x0, A_T, Bw_T, By_T, Cv_T, Dvw_T, Dvy_T, Cu_T, Duw_T, Duy_T, out);
  rinn_value_kernel<<<BT / 128, 256, 0, stream>>>(obs, log_stds, W0, b0, W1,
                                                  b1, W2, b2, out);
}
